// MyModel_87522843561185
// MI455X (gfx1250) — compile-verified, matrix operand order checked by reading, not run on hardware
//
#include <hip/hip_runtime.h>
#include <hip/hip_bf16.h>
#include <math.h>

#define VOCABSZ 10000
#define EMBD    256
#define UNITS   1024
#define BATCH   64
#define SEQT    256
#define H3      (3 * UNITS)

typedef __attribute__((ext_vector_type(16))) __bf16 v16bf;
typedef __attribute__((ext_vector_type(8)))  float  v8f;

union FragB16 { v16bf v; uint4 q[2]; };

__device__ __forceinline__ unsigned short f2bf(float f) {
  union { float f; unsigned int u; } c; c.f = f;
  unsigned int u = c.u;
  u += 0x7fffu + ((u >> 16) & 1u);   // round-to-nearest-even
  return (unsigned short)(u >> 16);
}

__device__ __forceinline__ float sigmoidf_(float x) {
  return 1.0f / (1.0f + __expf(-x));
}

// ---------------------------------------------------------------------------
// Embedding gather -> bf16 X[(t*B+b)][e]   (xs = transpose(emb[x],(1,0,2)))
// ---------------------------------------------------------------------------
__global__ void gather_embed(const int* __restrict__ x,
                             const float* __restrict__ emb,
                             unsigned short* __restrict__ X) {
  const long long total = (long long)SEQT * BATCH * EMBD;
  long long idx = (long long)blockIdx.x * blockDim.x + threadIdx.x;
  const long long stride = (long long)gridDim.x * blockDim.x;
  for (; idx < total; idx += stride) {
    int e = (int)(idx % EMBD);
    long long row = idx / EMBD;
    int b = (int)(row % BATCH);
    int t = (int)(row / BATCH);
    int tok = x[b * SEQT + t];
    X[idx] = f2bf(emb[(long long)tok * EMBD + e]);
  }
}

// ---------------------------------------------------------------------------
// Transpose + convert weights to bf16, N-major with contiguous K:
//   Ut[n][k] = bf16(U[k][n])  (3H x H),  Wt[n][e] = bf16(W[e][n])  (3H x E)
// ---------------------------------------------------------------------------
__global__ void conv_weights(const float* __restrict__ W,   // (EMBD, 3H)
                             const float* __restrict__ U,   // (UNITS, 3H)
                             unsigned short* __restrict__ Wt,
                             unsigned short* __restrict__ Ut) {
  const long long total = (long long)H3 * UNITS;
  long long idx = (long long)blockIdx.x * blockDim.x + threadIdx.x;
  const long long stride = (long long)gridDim.x * blockDim.x;
  for (; idx < total; idx += stride) {
    int k = (int)(idx % UNITS);
    int n = (int)(idx / UNITS);
    Ut[(size_t)n * UNITS + k] = f2bf(U[(size_t)k * H3 + n]);
    if (k < EMBD)   // EMBD < UNITS, so this covers all of Wt
      Wt[(size_t)n * EMBD + k] = f2bf(W[(size_t)k * H3 + n]);
  }
}

// ---------------------------------------------------------------------------
// Initialize recurrent state (fp32 master + bf16 matrix copy) for both dirs.
// ---------------------------------------------------------------------------
__global__ void init_h(const float* __restrict__ hidden,
                       float* __restrict__ hF, float* __restrict__ hB,
                       unsigned short* __restrict__ hFbf,
                       unsigned short* __restrict__ hBbf) {
  int total = BATCH * UNITS;
  for (int idx = blockIdx.x * blockDim.x + threadIdx.x; idx < total;
       idx += gridDim.x * blockDim.x) {
    float v = hidden[idx];
    unsigned short bv = f2bf(v);
    hF[idx] = v;  hB[idx] = v;  hFbf[idx] = bv;  hBbf[idx] = bv;
  }
}

// ---------------------------------------------------------------------------
// Phase 1 GEMM: xw = X @ W^T + b_i   -> fp32 (T*B, 3H)
// wave32: each wave computes one 16x16 tile; K-loop over E=256 in chunks of 32.
// ---------------------------------------------------------------------------
__global__ void __launch_bounds__(128)
xw_gemm(const unsigned short* __restrict__ X,   // (T*B, EMBD) bf16
        const unsigned short* __restrict__ Wt,  // (3H, EMBD)  bf16
        const float* __restrict__ bi,           // (3H)
        float* __restrict__ xw) {               // (T*B, 3H) fp32
  const int lane  = threadIdx.x & 31;
  const int wave  = threadIdx.x >> 5;
  const int l16   = lane & 15;
  const int half  = (lane >> 4) & 1;
  const int ntile = blockIdx.x;               // 0..191 over 3H
  const int mtile = blockIdx.y * 4 + wave;    // 0..1023 over T*B

  const unsigned short* aBase =
      X + (size_t)(mtile * 16 + l16) * EMBD + half * 8;
  const unsigned short* bBase =
      Wt + (size_t)(ntile * 16 + l16) * EMBD + half * 8;

  v8f acc = {};
  #pragma unroll
  for (int kb = 0; kb < EMBD; kb += 32) {
    FragB16 a, b;
    a.q[0] = *(const uint4*)(aBase + kb);
    a.q[1] = *(const uint4*)(aBase + kb + 16);
    b.q[0] = *(const uint4*)(bBase + kb);
    b.q[1] = *(const uint4*)(bBase + kb + 16);
    acc = __builtin_amdgcn_wmma_f32_16x16x32_bf16(
        false, a.v, false, b.v, (short)0, acc, false, false);
  }

  const int nOut = ntile * 16 + l16;
  const float bias = bi[nOut];
  #pragma unroll
  for (int i = 0; i < 8; ++i) {
    int m = mtile * 16 + half * 8 + i;
    xw[(size_t)m * H3 + nOut] = acc[i] + bias;
  }
}

// ---------------------------------------------------------------------------
// One GRU time step, both directions fused (blockIdx.z = dir).
// Block = 128 threads = 4 waves = 4 M-tiles (full B=64); grid.x = 64 H-tiles.
// Each wave: 3 accumulators (z, r, h-gate) over K=1024 -> 96 WMMA ops,
// then gate nonlinearity + state update + write into (B,T,2H) concat output.
// ---------------------------------------------------------------------------
__global__ void __launch_bounds__(128)
gru_step(const float* __restrict__ xwF, const float* __restrict__ xwB,
         const unsigned short* __restrict__ UtF,
         const unsigned short* __restrict__ UtB,
         const float* __restrict__ brF, const float* __restrict__ brB,
         const float* __restrict__ hFp, const float* __restrict__ hBp,
         const unsigned short* __restrict__ hFpbf,
         const unsigned short* __restrict__ hBpbf,
         float* __restrict__ hFn, float* __restrict__ hBn,
         unsigned short* __restrict__ hFnbf, unsigned short* __restrict__ hBnbf,
         float* __restrict__ out, int s) {
  const int dir = blockIdx.z;
  const float* xw            = dir ? xwB  : xwF;
  const unsigned short* Ut   = dir ? UtB  : UtF;
  const float* br            = dir ? brB  : brF;
  const float* hp            = dir ? hBp  : hFp;
  const unsigned short* hpbf = dir ? hBpbf : hFpbf;
  float* hn                  = dir ? hBn  : hFn;
  unsigned short* hnbf       = dir ? hBnbf : hFnbf;
  const int t_real  = dir ? (SEQT - 1 - s) : s;
  const int out_off = dir ? UNITS : 0;

  const int lane  = threadIdx.x & 31;
  const int wave  = threadIdx.x >> 5;   // m-tile 0..3
  const int l16   = lane & 15;
  const int half  = (lane >> 4) & 1;
  const int htile = blockIdx.x;         // 0..63 over H

  const int hcol = htile * 16 + l16;
  const unsigned short* aBase =
      hpbf + (size_t)(wave * 16 + l16) * UNITS + half * 8;
  const unsigned short* bz =
      Ut + (size_t)(0 * UNITS + hcol) * UNITS + half * 8;
  const unsigned short* brp =
      Ut + (size_t)(1 * UNITS + hcol) * UNITS + half * 8;
  const unsigned short* bh =
      Ut + (size_t)(2 * UNITS + hcol) * UNITS + half * 8;

  v8f az = {}, ar = {}, ah = {};
  for (int kb = 0; kb < UNITS; kb += 32) {
    FragB16 a, fz, fr, fh;
    a.q[0]  = *(const uint4*)(aBase + kb);
    a.q[1]  = *(const uint4*)(aBase + kb + 16);
    fz.q[0] = *(const uint4*)(bz + kb);
    fz.q[1] = *(const uint4*)(bz + kb + 16);
    fr.q[0] = *(const uint4*)(brp + kb);
    fr.q[1] = *(const uint4*)(brp + kb + 16);
    fh.q[0] = *(const uint4*)(bh + kb);
    fh.q[1] = *(const uint4*)(bh + kb + 16);
    az = __builtin_amdgcn_wmma_f32_16x16x32_bf16(
        false, a.v, false, fz.v, (short)0, az, false, false);
    ar = __builtin_amdgcn_wmma_f32_16x16x32_bf16(
        false, a.v, false, fr.v, (short)0, ar, false, false);
    ah = __builtin_amdgcn_wmma_f32_16x16x32_bf16(
        false, a.v, false, fh.v, (short)0, ah, false, false);
  }

  const float bz_s = br[hcol];
  const float br_s = br[UNITS + hcol];
  const float bh_s = br[2 * UNITS + hcol];
  const float* xwT = xw + (size_t)t_real * BATCH * H3;

  #pragma unroll
  for (int i = 0; i < 8; ++i) {
    const int m = wave * 16 + half * 8 + i;   // batch index
    const float* xr_ = xwT + (size_t)m * H3;
    const float xz = xr_[hcol];
    const float xr = xr_[UNITS + hcol];
    const float xh = xr_[2 * UNITS + hcol];
    const float z    = sigmoidf_(xz + az[i] + bz_s);
    const float r    = sigmoidf_(xr + ar[i] + br_s);
    const float cand = tanhf(xh + r * (ah[i] + bh_s));
    const float hprev = hp[(size_t)m * UNITS + hcol];
    const float hv = z * hprev + (1.0f - z) * cand;
    hn[(size_t)m * UNITS + hcol]   = hv;
    hnbf[(size_t)m * UNITS + hcol] = f2bf(hv);
    out[((size_t)m * SEQT + t_real) * (2 * UNITS) + out_off + hcol] = hv;
  }
}

// ---------------------------------------------------------------------------
// Copy final states hf, hb to the tail of d_out.
// ---------------------------------------------------------------------------
__global__ void final_h(const float* __restrict__ hF,
                        const float* __restrict__ hB,
                        float* __restrict__ out) {
  const size_t base = (size_t)BATCH * SEQT * 2 * UNITS;
  int total = BATCH * UNITS;
  for (int idx = blockIdx.x * blockDim.x + threadIdx.x; idx < total;
       idx += gridDim.x * blockDim.x) {
    out[base + idx] = hF[idx];
    out[base + BATCH * UNITS + idx] = hB[idx];
  }
}

// ---------------------------------------------------------------------------
extern "C" void kernel_launch(void* const* d_in, const int* in_sizes, int n_in,
                              void* d_out, int out_size, void* d_ws,
                              size_t ws_size, hipStream_t stream) {
  (void)in_sizes; (void)n_in; (void)out_size; (void)ws_size;
  const int*   x      = (const int*)d_in[0];
  const float* hidden = (const float*)d_in[1];
  const float* emb    = (const float*)d_in[2];
  const float* Wf     = (const float*)d_in[3];
  const float* Uf     = (const float*)d_in[4];
  const float* bf     = (const float*)d_in[5];   // [0]=b_i, [1]=b_r (each 3H)
  const float* Wb     = (const float*)d_in[6];
  const float* Ub     = (const float*)d_in[7];
  const float* bb     = (const float*)d_in[8];
  float* out = (float*)d_out;

  unsigned char* ws = (unsigned char*)d_ws;
  size_t off = 0;
  auto alloc = [&](size_t bytes) -> void* {
    off = (off + 255) & ~(size_t)255;
    void* p = ws + off;
    off += bytes;
    return p;
  };

  unsigned short* X    = (unsigned short*)alloc((size_t)SEQT * BATCH * EMBD * 2);
  unsigned short* WtF  = (unsigned short*)alloc((size_t)H3 * EMBD * 2);
  unsigned short* WtB  = (unsigned short*)alloc((size_t)H3 * EMBD * 2);
  unsigned short* UtF  = (unsigned short*)alloc((size_t)H3 * UNITS * 2);
  unsigned short* UtB  = (unsigned short*)alloc((size_t)H3 * UNITS * 2);
  float* xwF           = (float*)alloc((size_t)SEQT * BATCH * H3 * 4);
  float* xwB           = (float*)alloc((size_t)SEQT * BATCH * H3 * 4);
  float* hF[2]; float* hB[2];
  unsigned short* hFbf[2]; unsigned short* hBbf[2];
  for (int i = 0; i < 2; ++i) {
    hF[i]   = (float*)alloc((size_t)BATCH * UNITS * 4);
    hB[i]   = (float*)alloc((size_t)BATCH * UNITS * 4);
    hFbf[i] = (unsigned short*)alloc((size_t)BATCH * UNITS * 2);
    hBbf[i] = (unsigned short*)alloc((size_t)BATCH * UNITS * 2);
  }

  // Phase 0: gather + weight prep + state init
  gather_embed<<<2048, 256, 0, stream>>>(x, emb, X);
  conv_weights<<<4096, 256, 0, stream>>>(Wf, Uf, WtF, UtF);
  conv_weights<<<4096, 256, 0, stream>>>(Wb, Ub, WtB, UtB);
  init_h<<<256, 256, 0, stream>>>(hidden, hF[0], hB[0], hFbf[0], hBbf[0]);

  // Phase 1: big input-projection GEMMs (WMMA)
  dim3 g1(H3 / 16, (SEQT * BATCH) / (16 * 4), 1);
  xw_gemm<<<g1, 128, 0, stream>>>(X, WtF, bf /* b_i */, xwF);
  xw_gemm<<<g1, 128, 0, stream>>>(X, WtB, bb /* b_i */, xwB);

  // Phase 2: sequential recurrent scan, both directions fused per step
  dim3 g2(UNITS / 16, 1, 2);
  for (int s = 0; s < SEQT; ++s) {
    const int cur = s & 1, nxt = cur ^ 1;
    gru_step<<<g2, 128, 0, stream>>>(
        xwF, xwB, UtF, UtB, bf + H3, bb + H3,
        hF[cur], hB[cur], hFbf[cur], hBbf[cur],
        hF[nxt], hB[nxt], hFbf[nxt], hBbf[nxt],
        out, s);
  }

  // After 256 steps the final states land back in buffer 0.
  final_h<<<256, 256, 0, stream>>>(hF[0], hB[0], out);
}
